// SceneIQGAT_85255100826266
// MI455X (gfx1250) — compile-verified, matrix operand order checked by reading, not run on hardware
//
#include <hip/hip_runtime.h>
#include <math.h>

typedef __attribute__((ext_vector_type(16))) __bf16 v16bf;
typedef __attribute__((ext_vector_type(8)))  float  v8f;
typedef __attribute__((ext_vector_type(4)))  int    v4i_;
typedef __attribute__((address_space(1))) v4i_* gptr_v4i;   // global int4*
typedef __attribute__((address_space(3))) v4i_* lptr_v4i;   // LDS int4*

#ifndef __has_builtin
#define __has_builtin(x) 0
#endif
#if __has_builtin(__builtin_amdgcn_global_load_async_to_lds_b128) && \
    __has_builtin(__builtin_amdgcn_s_wait_asynccnt)
#define USE_ASYNC_LDS 1
#else
#define USE_ASYNC_LDS 0
#endif

#define BDIM 128
#define NDIM 128
#define MNODE (BDIM * NDIM)          // 16384 nodes
#define EEDGE 524288
#define ETOT  (EEDGE + MNODE)        // edges + self loops = 540672
#define DDIM 128
#define HDIM 256
#define NHEAD 4
#define CDIM 64
#define NLAYER 2
#define NOBJ 150
#define NPRED 50
#define NEG_SLOPE 0.2f

// ---------------------------------------------------------------- utilities

__device__ __forceinline__ void atomicMaxF(float* addr, float v) {
    unsigned int* ua = (unsigned int*)addr;
    unsigned int old = __float_as_uint(*addr);
    while (__uint_as_float(old) < v) {
        unsigned int assumed = old;
        old = atomicCAS(ua, assumed, __float_as_uint(v));
        if (old == assumed) break;
    }
}

__global__ void k_fill(float* p, float v, int n) {
    int i = blockIdx.x * blockDim.x + threadIdx.x;
    if (i < n) p[i] = v;
}

__global__ void k_cvt_bf16(const float* __restrict__ s, __bf16* __restrict__ d, int n) {
    int i = blockIdx.x * blockDim.x + threadIdx.x;
    if (i < n) d[i] = (__bf16)s[i];
}

// WT[c * rows + r] = W[r * cols + c]  (transpose + convert to bf16)
__global__ void k_transpose_bf16(const float* __restrict__ W, __bf16* __restrict__ WT,
                                 int rows, int cols) {
    int i = blockIdx.x * blockDim.x + threadIdx.x;
    if (i >= rows * cols) return;
    int r = i / cols, c = i % cols;
    WT[(size_t)c * rows + r] = (__bf16)W[(size_t)r * cols + c];
}

// V[l][d][h] = sum_c W_edge[l,d,h*64+c] * att_edge[l,h,c]
__global__ void k_vtab(const float* __restrict__ W_edge, const float* __restrict__ att_edge,
                       float* __restrict__ V) {
    int i = blockIdx.x * blockDim.x + threadIdx.x;
    if (i >= NLAYER * DDIM * NHEAD) return;
    int h = i % NHEAD, d = (i / NHEAD) % DDIM, l = i / (NHEAD * DDIM);
    const float* we = W_edge + ((size_t)l * DDIM + d) * HDIM + h * CDIM;
    const float* ae = att_edge + ((size_t)l * NHEAD + h) * CDIM;
    float s = 0.f;
    for (int c = 0; c < CDIM; ++c) s += we[c] * ae[c];
    V[i] = s;
}

// T[l][p][h] = sum_d pred_emb[p,d] * V[l,d,h]
__global__ void k_ttab(const float* __restrict__ pred_emb, const float* __restrict__ V,
                       float* __restrict__ T) {
    int i = blockIdx.x * blockDim.x + threadIdx.x;
    if (i >= NLAYER * NPRED * NHEAD) return;
    int h = i % NHEAD, p = (i / NHEAD) % NPRED, l = i / (NHEAD * NPRED);
    float s = 0.f;
    for (int d = 0; d < DDIM; ++d)
        s += pred_emb[(size_t)p * DDIM + d] * V[((size_t)l * DDIM + d) * NHEAD + h];
    T[i] = s;
}

// ------------------------------------------------- WMMA GEMM (LDS-staged B)
// out = [relu](A[gather] @ BT^T + bias). A: M x K bf16, BT: 256 x K bf16 (pre-transposed).
// grid (M/128, 256/64), block 256 = 8 waves; each wave: 16 rows x 64 cols (4 tiles).
// B tile (64 cols x K) staged once per block into LDS via async copy (ASYNCcnt path).
template <int K, bool GATHER>
__global__ void k_gemm_bf16(const int* __restrict__ ids, const __bf16* __restrict__ A,
                            const __bf16* __restrict__ BT, const float* __restrict__ bias,
                            float* __restrict__ out_f32, __bf16* __restrict__ out_bf16) {
    __shared__ __attribute__((aligned(16))) __bf16 Bs[64 * K];
    int lane = threadIdx.x & 31, wv = threadIdx.x >> 5;
    int half = lane >> 4, lr = lane & 15;
    int i0 = (blockIdx.x * 8 + wv) * 16;
    int j0 = blockIdx.y * 64;

    // stage B tile (contiguous 64*K bf16 block of BT) into LDS
    const __bf16* gsrc = BT + (size_t)j0 * K;
#if USE_ASYNC_LDS
    for (int c = threadIdx.x * 8; c < 64 * K; c += 256 * 8) {
        __builtin_amdgcn_global_load_async_to_lds_b128(
            (gptr_v4i)(gsrc + c), (lptr_v4i)(Bs + c), 0, 0);
    }
    __builtin_amdgcn_s_wait_asynccnt(0);
#else
    for (int c = threadIdx.x * 8; c < 64 * K; c += 256 * 8)
        *(int4*)(Bs + c) = *(const int4*)(gsrc + c);
#endif
    __syncthreads();

    const __bf16* arow = A + (size_t)(GATHER ? ids[i0 + lr] : (i0 + lr)) * K;
    v8f acc[4] = {};
    for (int k0 = 0; k0 < K; k0 += 32) {
        v16bf a = *(const v16bf*)(arow + k0 + half * 16);
#pragma unroll
        for (int j = 0; j < 4; ++j) {
            const __bf16* bp2 = Bs + (size_t)(j * 16 + lr) * K + k0 + half * 16;
            v16bf b = *(const v16bf*)bp2;
            acc[j] = __builtin_amdgcn_wmma_f32_16x16x32_bf16(
                false, a, false, b, (short)0, acc[j], false, false);
        }
    }
#pragma unroll
    for (int j = 0; j < 4; ++j) {
        int col = j0 + j * 16 + lr;
        float bv = bias ? bias[col] : 0.f;
#pragma unroll
        for (int r = 0; r < 8; ++r) {
            int orow = i0 + r + half * 8;
            float v = acc[j][r] + bv;
            if (out_f32) {                       // projection path: relu + fp32 copy
                v = v > 0.f ? v : 0.f;
                out_f32[(size_t)orow * HDIM + col] = v;
            }
            out_bf16[(size_t)orow * HDIM + col] = (__bf16)v;
        }
    }
}

// a_src[m,h] = sum_c xs[m,h,c]*att_src[h,c]; same for a_dst. One block per node.
__global__ void k_attn_scores(const __bf16* __restrict__ xs, const float* __restrict__ as,
                              const float* __restrict__ ad, float* __restrict__ a_src,
                              float* __restrict__ a_dst) {
    __shared__ float ss[HDIM], sd[HDIM];
    int m = blockIdx.x, t = threadIdx.x;
    float x = (float)xs[(size_t)m * HDIM + t];
    ss[t] = x * as[t];
    sd[t] = x * ad[t];
    __syncthreads();
    int c = t & 63;
    for (int off = 32; off > 0; off >>= 1) {
        if (c < off) { ss[t] += ss[t + off]; sd[t] += sd[t + off]; }
        __syncthreads();
    }
    if (c == 0) {
        int h = t >> 6;
        a_src[m * NHEAD + h] = ss[t];
        a_dst[m * NHEAD + h] = sd[t];
    }
}

// per real edge: deg[dst]++, sumT{0,1}[dst,h] += T{0,1}[attr,h]
__global__ void k_deg_sumT(const int* __restrict__ ei, const int* __restrict__ attr,
                           const float* __restrict__ T0, const float* __restrict__ T1,
                           float* __restrict__ deg, float* __restrict__ s0,
                           float* __restrict__ s1) {
    int i = blockIdx.x * blockDim.x + threadIdx.x;
    if (i >= EEDGE) return;
    int d = ei[EEDGE + i];
    int a = attr[i];
    atomicAdd(&deg[d], 1.f);
#pragma unroll
    for (int h = 0; h < NHEAD; ++h) {
        atomicAdd(&s0[d * NHEAD + h], T0[a * NHEAD + h]);
        atomicAdd(&s1[d * NHEAD + h], T1[a * NHEAD + h]);
    }
}

// pass 1: alpha = leaky(a_src[s]+a_dst[d]+a_edg), atomic-max per (dst,h)
__global__ void k_alpha_max(const int* __restrict__ ei, const int* __restrict__ attr,
                            const float* __restrict__ T, const float* __restrict__ sumT,
                            const float* __restrict__ deg, const float* __restrict__ a_src,
                            const float* __restrict__ a_dst, float* __restrict__ alpha,
                            float* __restrict__ amax) {
    int tid = blockIdx.x * blockDim.x + threadIdx.x;
    if (tid >= ETOT * NHEAD) return;
    int i = tid >> 2, h = tid & 3;
    int s, d;
    float aedg;
    if (i < EEDGE) {
        s = ei[i]; d = ei[EEDGE + i];
        aedg = T[attr[i] * NHEAD + h];
    } else {
        s = d = i - EEDGE;
        float dg = deg[d];
        aedg = sumT[d * NHEAD + h] / (dg > 1.f ? dg : 1.f);
    }
    float a = a_src[s * NHEAD + h] + a_dst[d * NHEAD + h] + aedg;
    a = a > 0.f ? a : NEG_SLOPE * a;
    alpha[tid] = a;
    atomicMaxF(&amax[d * NHEAD + h], a);
}

// pass 2: ex = exp(alpha - amax[dst]); den[dst] += ex (alpha overwritten with ex)
__global__ void k_exp_den(const int* __restrict__ ei, float* __restrict__ alpha,
                          const float* __restrict__ amax, float* __restrict__ den) {
    int tid = blockIdx.x * blockDim.x + threadIdx.x;
    if (tid >= ETOT * NHEAD) return;
    int i = tid >> 2, h = tid & 3;
    int d = (i < EEDGE) ? ei[EEDGE + i] : (i - EEDGE);
    float am = amax[d * NHEAD + h];
    if (am < -1e29f) am = 0.f;                 // mirror isfinite guard
    float ex = __expf(alpha[tid] - am);
    alpha[tid] = ex;
    atomicAdd(&den[d * NHEAD + h], ex);
}

// pass 3: agg[dst,:] += xs[src,:] * ex/(den[dst]+eps). 64 threads per edge, 4 ch each.
__global__ void k_aggregate(const int* __restrict__ ei, const float* __restrict__ alpha,
                            const float* __restrict__ den, const __bf16* __restrict__ xs,
                            float* __restrict__ agg) {
    unsigned int tid = blockIdx.x * blockDim.x + threadIdx.x;
    unsigned int i = tid >> 6;
    if (i >= ETOT) return;
    int t = tid & 63;
    int s, d;
    if (i < EEDGE) { s = ei[i]; d = ei[EEDGE + i]; }
    else           { s = d = (int)i - EEDGE; }
    int h = t >> 4;
    float w = alpha[(size_t)i * NHEAD + h] / (den[d * NHEAD + h] + 1e-16f);
    const __bf16* xp = xs + (size_t)s * HDIM + t * 4;
    float* ap = agg + (size_t)d * HDIM + t * 4;
    atomicAdd(ap + 0, (float)xp[0] * w);
    atomicAdd(ap + 1, (float)xp[1] * w);
    atomicAdd(ap + 2, (float)xp[2] * w);
    atomicAdd(ap + 3, (float)xp[3] * w);
}

// h = elu(agg + b_gat) + flat; LayerNorm(H); write flat_f32 + flat_bf16 (one block/node)
__global__ void k_finalize_ln(const float* __restrict__ agg, const float* __restrict__ bg,
                              const float* __restrict__ g, const float* __restrict__ b,
                              float* __restrict__ flat, __bf16* __restrict__ flat_bf) {
    __shared__ float red[HDIM];
    __shared__ float smu, svar;
    int m = blockIdx.x, t = threadIdx.x;
    float a = agg[(size_t)m * HDIM + t] + bg[t];
    float e = a > 0.f ? a : (__expf(a) - 1.f);
    float hv = e + flat[(size_t)m * HDIM + t];
    red[t] = hv;
    __syncthreads();
    for (int off = 128; off > 0; off >>= 1) {
        if (t < off) red[t] += red[t + off];
        __syncthreads();
    }
    if (t == 0) smu = red[0] * (1.f / HDIM);
    __syncthreads();
    float dv = hv - smu;
    red[t] = dv * dv;
    __syncthreads();
    for (int off = 128; off > 0; off >>= 1) {
        if (t < off) red[t] += red[t + off];
        __syncthreads();
    }
    if (t == 0) svar = red[0] * (1.f / HDIM);
    __syncthreads();
    float o = g[t] * dv * rsqrtf(svar + 1e-5f) + b[t];
    flat[(size_t)m * HDIM + t] = o;
    flat_bf[(size_t)m * HDIM + t] = (__bf16)o;
}

// graph mean-pool + scs logit. One block per batch element.
__global__ void k_output(const float* __restrict__ flat, const unsigned char* __restrict__ mask,
                         const float* __restrict__ Ws, const float* __restrict__ bs,
                         float* __restrict__ out) {
    __shared__ float red[HDIM];
    int bb = blockIdx.x, t = threadIdx.x;
    float s = 0.f, cnt = 0.f;
    for (int n = 0; n < NDIM; ++n) {
        float mk = mask[bb * NDIM + n] ? 1.f : 0.f;
        cnt += mk;
        s += flat[((size_t)(bb * NDIM + n)) * HDIM + t] * mk;
    }
    float ge = s / (cnt > 1.f ? cnt : 1.f);
    out[BDIM + bb * HDIM + t] = ge;               // graph_emb section
    red[t] = ge * Ws[t];
    __syncthreads();
    for (int off = 128; off > 0; off >>= 1) {
        if (t < off) red[t] += red[t + off];
        __syncthreads();
    }
    if (t == 0) out[bb] = red[0] + bs[0];         // scs_logit section
}

// copy nodes (flat) and mask (as 0/1 float) into d_out tail sections
__global__ void k_copy_out(const float* __restrict__ flat, const unsigned char* __restrict__ mask,
                           float* __restrict__ out) {
    size_t i = (size_t)blockIdx.x * blockDim.x + threadIdx.x;
    const size_t nn = (size_t)MNODE * HDIM;
    if (i < nn) out[BDIM + BDIM * HDIM + i] = flat[i];
    else if (i < nn + MNODE) {
        size_t j = i - nn;
        out[BDIM + BDIM * HDIM + nn + j] = mask[j] ? 1.f : 0.f;
    }
}

// ---------------------------------------------------------------- host

extern "C" void kernel_launch(void* const* d_in, const int* in_sizes, int n_in,
                              void* d_out, int out_size, void* d_ws, size_t ws_size,
                              hipStream_t stream) {
    const int* object_ids = (const int*)d_in[0];
    const unsigned char* node_mask = (const unsigned char*)d_in[1];
    const int* edge_index = (const int*)d_in[2];
    const int* edge_attr = (const int*)d_in[3];
    const float* obj_emb = (const float*)d_in[4];
    const float* pred_emb = (const float*)d_in[5];
    const float* Wp = (const float*)d_in[6];
    const float* bp = (const float*)d_in[7];
    const float* W_gat = (const float*)d_in[8];
    const float* W_edge = (const float*)d_in[9];
    const float* att_src = (const float*)d_in[10];
    const float* att_dst = (const float*)d_in[11];
    const float* att_edge = (const float*)d_in[12];
    const float* b_gat = (const float*)d_in[13];
    const float* ln_g = (const float*)d_in[14];
    const float* ln_b = (const float*)d_in[15];
    const float* Ws = (const float*)d_in[16];
    const float* bs = (const float*)d_in[17];
    float* out = (float*)d_out;

    // workspace carve-up (256B aligned)
    char* base = (char*)d_ws;
    size_t off = 0;
    auto carve = [&](size_t bytes) { size_t o = off; off += (bytes + 255) & ~(size_t)255; return o; };
    __bf16* obj_bf  = (__bf16*)(base + carve((size_t)NOBJ * DDIM * 2));
    __bf16* WpT     = (__bf16*)(base + carve((size_t)HDIM * DDIM * 2));
    __bf16* WgatT   = (__bf16*)(base + carve((size_t)NLAYER * HDIM * HDIM * 2));
    float*  Vtab    = (float*)(base + carve((size_t)NLAYER * DDIM * NHEAD * 4));
    float*  Ttab    = (float*)(base + carve((size_t)NLAYER * NPRED * NHEAD * 4));
    float*  flat    = (float*)(base + carve((size_t)MNODE * HDIM * 4));
    __bf16* flat_bf = (__bf16*)(base + carve((size_t)MNODE * HDIM * 2));
    __bf16* xs_bf   = (__bf16*)(base + carve((size_t)MNODE * HDIM * 2));
    float*  a_src   = (float*)(base + carve((size_t)MNODE * NHEAD * 4));
    float*  a_dst   = (float*)(base + carve((size_t)MNODE * NHEAD * 4));
    float*  deg     = (float*)(base + carve((size_t)MNODE * 4));
    float*  sumT    = (float*)(base + carve((size_t)NLAYER * MNODE * NHEAD * 4));
    float*  alpha   = (float*)(base + carve((size_t)ETOT * NHEAD * 4));
    float*  amax    = (float*)(base + carve((size_t)MNODE * NHEAD * 4));
    float*  den     = (float*)(base + carve((size_t)MNODE * NHEAD * 4));
    float*  agg     = (float*)(base + carve((size_t)MNODE * HDIM * 4));
    (void)ws_size; (void)n_in; (void)in_sizes; (void)out_size;

    // weights -> bf16 (B pre-transposed so WMMA B-fragments are contiguous)
    k_cvt_bf16<<<(NOBJ * DDIM + 255) / 256, 256, 0, stream>>>(obj_emb, obj_bf, NOBJ * DDIM);
    k_transpose_bf16<<<(DDIM * HDIM + 255) / 256, 256, 0, stream>>>(Wp, WpT, DDIM, HDIM);
    for (int l = 0; l < NLAYER; ++l)
        k_transpose_bf16<<<(HDIM * HDIM + 255) / 256, 256, 0, stream>>>(
            W_gat + (size_t)l * HDIM * HDIM, WgatT + (size_t)l * HDIM * HDIM, HDIM, HDIM);
    // collapse W_edge/att_edge path to tiny tables (a_edg = T[edge_attr])
    k_vtab<<<(NLAYER * DDIM * NHEAD + 255) / 256, 256, 0, stream>>>(W_edge, att_edge, Vtab);
    k_ttab<<<(NLAYER * NPRED * NHEAD + 255) / 256, 256, 0, stream>>>(pred_emb, Vtab, Ttab);

    // x = relu(gather @ Wp + bp)  -- WMMA, async-LDS staged B
    k_gemm_bf16<DDIM, true><<<dim3(MNODE / 128, HDIM / 64), 256, 0, stream>>>(
        object_ids, obj_bf, WpT, bp, flat, flat_bf);

    // deg and per-dst sums of T (for self-loop mean edge attr)
    (void)hipMemsetAsync(deg, 0, (size_t)MNODE * 4, stream);
    (void)hipMemsetAsync(sumT, 0, (size_t)NLAYER * MNODE * NHEAD * 4, stream);
    k_deg_sumT<<<EEDGE / 256, 256, 0, stream>>>(
        edge_index, edge_attr, Ttab, Ttab + NPRED * NHEAD, deg, sumT, sumT + MNODE * NHEAD);

    for (int l = 0; l < NLAYER; ++l) {
        const float* Tl = Ttab + (size_t)l * NPRED * NHEAD;
        float* sTl = sumT + (size_t)l * MNODE * NHEAD;
        // xs = flat @ W_gat[l]  -- WMMA, async-LDS staged B
        k_gemm_bf16<HDIM, false><<<dim3(MNODE / 128, HDIM / 64), 256, 0, stream>>>(
            nullptr, flat_bf, WgatT + (size_t)l * HDIM * HDIM, nullptr, nullptr, xs_bf);
        k_attn_scores<<<MNODE, HDIM, 0, stream>>>(
            xs_bf, att_src + (size_t)l * HDIM, att_dst + (size_t)l * HDIM, a_src, a_dst);
        k_fill<<<(MNODE * NHEAD + 255) / 256, 256, 0, stream>>>(amax, -1e30f, MNODE * NHEAD);
        (void)hipMemsetAsync(den, 0, (size_t)MNODE * NHEAD * 4, stream);
        (void)hipMemsetAsync(agg, 0, (size_t)MNODE * HDIM * 4, stream);
        k_alpha_max<<<(ETOT * NHEAD) / 256, 256, 0, stream>>>(
            edge_index, edge_attr, Tl, sTl, deg, a_src, a_dst, alpha, amax);
        k_exp_den<<<(ETOT * NHEAD) / 256, 256, 0, stream>>>(edge_index, alpha, amax, den);
        k_aggregate<<<ETOT / 4, 256, 0, stream>>>(edge_index, alpha, den, xs_bf, agg);
        k_finalize_ln<<<MNODE, HDIM, 0, stream>>>(
            agg, b_gat + (size_t)l * HDIM, ln_g + (size_t)l * HDIM, ln_b + (size_t)l * HDIM,
            flat, flat_bf);
    }

    k_output<<<BDIM, HDIM, 0, stream>>>(flat, node_mask, Ws, bs, out);
    {
        size_t tot = (size_t)MNODE * HDIM + MNODE;
        k_copy_out<<<(unsigned)((tot + 255) / 256), 256, 0, stream>>>(flat, node_mask, out);
    }
}
